// BertSelfAttention10_59236188946620
// MI455X (gfx1250) — compile-verified, matrix operand order checked by reading, not run on hardware
//
#include <hip/hip_runtime.h>

typedef __attribute__((ext_vector_type(16))) __bf16 v16bf;
typedef __attribute__((ext_vector_type(8)))  __bf16 v8bf;
typedef __attribute__((ext_vector_type(8)))  float  v8f;

#define NB   2
#define SEQ  4096
#define HD   768
#define NH   12
#define DH   64

// ---- optional CDNA5 async global->LDS path (guarded; falls back cleanly) ----
#if defined(__has_builtin)
# if __has_builtin(__builtin_amdgcn_global_load_async_to_lds_b128)
#  define USE_ASYNC_LDS 1
# endif
#endif
#ifndef USE_ASYNC_LDS
# define USE_ASYNC_LDS 0
#endif

// Builtin params are int4* in AS(1)/AS(3) per hipcc diagnostic.
typedef int v4i_t __attribute__((__vector_size__(16)));
typedef __attribute__((address_space(1))) v4i_t as1_v4i;
typedef __attribute__((address_space(3))) v4i_t as3_v4i;

static __device__ __forceinline__ void cp16(__bf16* dst_lds, const __bf16* src_g) {
#if USE_ASYNC_LDS
    __builtin_amdgcn_global_load_async_to_lds_b128(
        (as1_v4i*)(src_g), (as3_v4i*)(dst_lds), 0, 0);
#else
    *(v8bf*)dst_lds = *(const v8bf*)src_g;
#endif
}

static __device__ __forceinline__ void cp_wait() {
#if USE_ASYNC_LDS
# if __has_builtin(__builtin_amdgcn_s_wait_asynccnt)
    __builtin_amdgcn_s_wait_asynccnt(0);
# else
    asm volatile("s_wait_asynccnt 0x0" ::: "memory");
# endif
#endif
}

// round-to-nearest-even fp32 -> bf16
static __device__ __forceinline__ __bf16 f2bf(float f) {
    union { float f; unsigned u; } c; c.f = f;
    unsigned r = c.u + 0x7FFFu + ((c.u >> 16) & 1u);
    unsigned short hs = (unsigned short)(r >> 16);
    union { unsigned short s; __bf16 b; } o; o.s = hs; return o.b;
}

// Load a 16x32 bf16 WMMA A/B fragment from a K-contiguous row/column.
// CDNA5 layout: elements 0..7 = K[half*8 .. +7], 8..15 = K[16+half*8 .. +7].
static __device__ __forceinline__ v16bf load_frag(const __bf16* row, int half) {
    v16bf f;
    ((v8bf*)&f)[0] = *(const v8bf*)(row + half * 8);
    ((v8bf*)&f)[1] = *(const v8bf*)(row + 16 + half * 8);
    return f;
}

static __device__ __forceinline__ v8f wmma_bf16(v16bf a, v16bf b, v8f c) {
    return __builtin_amdgcn_wmma_f32_16x16x32_bf16(false, a, false, b, (short)0, c, false, false);
}

// ---------------- fp32 -> bf16 convert ----------------
__global__ __launch_bounds__(256) void cvt_bf16(const float* __restrict__ in,
                                                __bf16* __restrict__ out, int n) {
    int i = (blockIdx.x * 256 + threadIdx.x) * 4;
    if (i + 3 < n) {
        float4 v = *(const float4*)(in + i);
        out[i + 0] = f2bf(v.x);
        out[i + 1] = f2bf(v.y);
        out[i + 2] = f2bf(v.z);
        out[i + 3] = f2bf(v.w);
    }
}

// ---------------- fused QKV projection ----------------
__global__ __launch_bounds__(256) void qkv_proj(
    const __bf16* __restrict__ X,
    const __bf16* __restrict__ Wq, const __bf16* __restrict__ Wk, const __bf16* __restrict__ Wv,
    const float* __restrict__ bq, const float* __restrict__ bk, const float* __restrict__ bv,
    __bf16* __restrict__ Qo,   // [B*S, HD], holds (q + bq)/8
    __bf16* __restrict__ Km,   // [B*S, HD], holds k_mod
    __bf16* __restrict__ Vt)   // [B, HD, SEQ] transposed V
{
    const int lane = threadIdx.x & 31;
    const int gw   = blockIdx.x * 8 + (threadIdx.x >> 5);
    const int stile = gw / (HD / 16);
    const int ntile = gw % (HD / 16);
    const int l15 = lane & 15, half = lane >> 4;

    const __bf16* xrow = X  + (size_t)(stile * 16 + l15) * HD;
    const __bf16* wqr  = Wq + (size_t)(ntile * 16 + l15) * HD;
    const __bf16* wkr  = Wk + (size_t)(ntile * 16 + l15) * HD;
    const __bf16* wvr  = Wv + (size_t)(ntile * 16 + l15) * HD;

    v8f accQ = {}, accK = {}, accVT = {}, accVN = {};
#pragma unroll 4
    for (int k0 = 0; k0 < HD; k0 += 32) {
        v16bf xf = load_frag(xrow + k0, half);
        v16bf qf = load_frag(wqr + k0, half);
        v16bf kf = load_frag(wkr + k0, half);
        v16bf vf = load_frag(wvr + k0, half);
        accQ  = wmma_bf16(qf, xf, accQ);   // D[n, s]
        accK  = wmma_bf16(kf, xf, accK);   // D[n, s]
        accVT = wmma_bf16(vf, xf, accVT);  // D[n, s]  (for k_mod)
        accVN = wmma_bf16(xf, vf, accVN);  // D[s, n]  (for Vt)
    }

    // ---- T-fragment epilogue: lane = column s, regs = 8 consecutive n ----
    const int ncol0 = ntile * 16 + 8 * half;
    float4 q0 = *(const float4*)(bq + ncol0), q1 = *(const float4*)(bq + ncol0 + 4);
    float4 k0v = *(const float4*)(bk + ncol0), k1v = *(const float4*)(bk + ncol0 + 4);
    float4 v0 = *(const float4*)(bv + ncol0), v1 = *(const float4*)(bv + ncol0 + 4);
    float bqa[8] = {q0.x, q0.y, q0.z, q0.w, q1.x, q1.y, q1.z, q1.w};
    float bka[8] = {k0v.x, k0v.y, k0v.z, k0v.w, k1v.x, k1v.y, k1v.z, k1v.w};
    float bva[8] = {v0.x, v0.y, v0.z, v0.w, v1.x, v1.y, v1.z, v1.w};

    const int srowT = stile * 16 + l15;
    v8bf qpack, kpack;
#pragma unroll
    for (int g = 0; g < 8; ++g) {
        float qv = (accQ[g] + bqa[g]) * 0.125f;            // fold 1/sqrt(D)
        float kv = accK[g] + bka[g];
        float vv = accVT[g] + bva[g];
        float s1 = kv / (1.0f + fabsf(kv));                // softsign(k)
        float s2 = s1 * 0.125f;
        float s3 = s2 / (1.0f + fabsf(s2));                // softsign(softsign(k)/dh)
        float km = kv * 0.125f + s3 + vv;                  // k/dh + ... + v
        qpack[g] = f2bf(qv);
        kpack[g] = f2bf(km);
    }
    *(v8bf*)(Qo + (size_t)srowT * HD + ncol0) = qpack;
    *(v8bf*)(Km + (size_t)srowT * HD + ncol0) = kpack;

    // ---- N-fragment epilogue (Vt): lane = column d, regs = 8 consecutive s ----
    const int dg    = ntile * 16 + l15;
    const int sbase = stile * 16 + 8 * half;
    const int bb    = sbase / SEQ;
    const int sin   = sbase % SEQ;
    const float bvn = bv[dg];
    v8bf vpack;
#pragma unroll
    for (int g = 0; g < 8; ++g) vpack[g] = f2bf(accVN[g] + bvn);
    *(v8bf*)(Vt + ((size_t)(bb * HD + dg)) * SEQ + sin) = vpack;
}

// ---------------- flash attention (LDS-shared K/V blocks) ----------------
// Workgroup = 8 waves, all same (b,h); wave w owns 16 queries. Each 64-key
// block of K_mod and V^T is staged to LDS once (async-to-LDS when available)
// and consumed by all 8 waves -> 8x less L2 traffic.
__global__ __launch_bounds__(256) void attn(
    const __bf16* __restrict__ Qo, const __bf16* __restrict__ Km,
    const __bf16* __restrict__ Vt, const int* __restrict__ mask,
    float* __restrict__ out)
{
    __shared__ __bf16 sK[64 * 64];   // [key][d]  row stride 64
    __shared__ __bf16 sV[64 * 64];   // [d][key]  row stride 64

    const int tid  = threadIdx.x;
    const int lane = tid & 31, wid = tid >> 5;
    const int nQB  = SEQ / 128;                    // 32 query-blocks of 128
    const int qb   = blockIdx.x % nQB;
    const int bh   = blockIdx.x / nQB;
    const int h    = bh % NH;
    const int b    = bh / NH;
    const int qt   = qb * 8 + wid;                 // this wave's 16-query tile
    const int l15 = lane & 15, half = lane >> 4;
    const float LOG2E = 1.4426950408889634f;

    const size_t qrow = (size_t)(b * SEQ + qt * 16 + l15);
    const __bf16* qbase = Qo + qrow * HD + h * DH;
    v16bf qf0 = load_frag(qbase,      half);       // d = 0..31
    v16bf qf1 = load_frag(qbase + 32, half);       // d = 32..63
    const int mq = mask[b * SEQ + qt * 16 + l15];

    const __bf16* kmbase = Km + (size_t)b * SEQ * HD + h * DH;      // + key*HD
    const __bf16* vtbase = Vt + (size_t)(b * HD + h * DH) * SEQ;    // + d*SEQ + key

    float m = -3.0e38f, l = 0.0f;
    v8f o[4] = {};

    for (int k0 = 0; k0 < SEQ; k0 += 64) {
        __syncthreads();   // previous block fully consumed before overwrite

        // ---- stage 8KB K_mod + 8KB V^T: 1024 16B units, 4 per thread ----
        {
            const __bf16* kmblk = kmbase + (size_t)k0 * HD;
            const __bf16* vtblk = vtbase + k0;
            int u0 = tid, u1 = tid + 256;
            int r0 = u0 >> 3, s0 = (u0 & 7) * 8;
            int r1 = u1 >> 3, s1 = (u1 & 7) * 8;
            cp16(sK + r0 * 64 + s0, kmblk + (size_t)r0 * HD + s0);
            cp16(sK + r1 * 64 + s1, kmblk + (size_t)r1 * HD + s1);
            cp16(sV + r0 * 64 + s0, vtblk + (size_t)r0 * SEQ + s0);
            cp16(sV + r1 * 64 + s1, vtblk + (size_t)r1 * SEQ + s1);
        }
        cp_wait();
        __syncthreads();

        // ---- scores (transposed): S^T[key, q] for 4 key-tiles of 16 ----
        v8f sc[4];
#pragma unroll
        for (int j = 0; j < 4; ++j) {
            const __bf16* kr = sK + (j * 16 + l15) * 64;
            v16bf f0 = load_frag(kr,      half);
            v16bf f1 = load_frag(kr + 32, half);
            v8f acc = {};
            acc = wmma_bf16(f0, qf0, acc);
            acc = wmma_bf16(f1, qf1, acc);
            sc[j] = acc;
        }

        // ---- online softmax over the 64-key block ----
        float p[4][8];
        float bm = -3.0e38f;
#pragma unroll
        for (int j = 0; j < 4; ++j)
#pragma unroll
            for (int g = 0; g < 8; ++g) {
                float a = mq ? sc[j][g] : -10000.0f;   // mask along QUERY dim
                p[j][g] = a;
                bm = fmaxf(bm, a);
            }
        bm = fmaxf(bm, __shfl_xor(bm, 16, 32));
        float mn   = fmaxf(m, bm);
        float corr = __builtin_amdgcn_exp2f((m - mn) * LOG2E);

        float lsum = 0.0f;
        v16bf pf0, pf1;                 // P^T as PV-WMMA B fragments
#pragma unroll
        for (int g = 0; g < 8; ++g) {
            float e0 = __builtin_amdgcn_exp2f((p[0][g] - mn) * LOG2E);
            float e1 = __builtin_amdgcn_exp2f((p[1][g] - mn) * LOG2E);
            float e2 = __builtin_amdgcn_exp2f((p[2][g] - mn) * LOG2E);
            float e3 = __builtin_amdgcn_exp2f((p[3][g] - mn) * LOG2E);
            lsum += (e0 + e1) + (e2 + e3);
            pf0[g] = f2bf(e0);  pf0[8 + g] = f2bf(e1);
            pf1[g] = f2bf(e2);  pf1[8 + g] = f2bf(e3);
        }
        lsum += __shfl_xor(lsum, 16, 32);
        l = l * corr + lsum;
#pragma unroll
        for (int dt = 0; dt < 4; ++dt) o[dt] = o[dt] * corr;

        // ---- O^T += V^T x P^T ----
#pragma unroll
        for (int dt = 0; dt < 4; ++dt) {
            const __bf16* vr = sV + (dt * 16 + l15) * 64;
            o[dt] = wmma_bf16(load_frag(vr,      half), pf0, o[dt]);
            o[dt] = wmma_bf16(load_frag(vr + 32, half), pf1, o[dt]);
        }

        m = mn;
    }

    const float linv = 1.0f / l;
    float* orow = out + qrow * HD + h * DH + 8 * half;
#pragma unroll
    for (int dt = 0; dt < 4; ++dt) {
        float4 lo = {o[dt][0] * linv, o[dt][1] * linv, o[dt][2] * linv, o[dt][3] * linv};
        float4 hi = {o[dt][4] * linv, o[dt][5] * linv, o[dt][6] * linv, o[dt][7] * linv};
        *(float4*)(orow + dt * 16)     = lo;
        *(float4*)(orow + dt * 16 + 4) = hi;
    }
}

extern "C" void kernel_launch(void* const* d_in, const int* in_sizes, int n_in,
                              void* d_out, int out_size, void* d_ws, size_t ws_size,
                              hipStream_t stream) {
    (void)in_sizes; (void)n_in; (void)out_size; (void)ws_size;
    const float* hs  = (const float*)d_in[0];
    const int*   msk = (const int*)d_in[1];
    const float* Wq  = (const float*)d_in[2];
    const float* bq  = (const float*)d_in[3];
    const float* Wk  = (const float*)d_in[4];
    const float* bk  = (const float*)d_in[5];
    const float* Wv  = (const float*)d_in[6];
    const float* bv  = (const float*)d_in[7];
    float* out = (float*)d_out;

    char* ws = (char*)d_ws;
    size_t off = 0;
    const size_t nX = (size_t)NB * SEQ * HD;   // 6,291,456
    const size_t nW = (size_t)HD * HD;         // 589,824
    __bf16* Xb  = (__bf16*)(ws + off); off += nX * 2;
    __bf16* Wqb = (__bf16*)(ws + off); off += nW * 2;
    __bf16* Wkb = (__bf16*)(ws + off); off += nW * 2;
    __bf16* Wvb = (__bf16*)(ws + off); off += nW * 2;
    __bf16* Qo  = (__bf16*)(ws + off); off += nX * 2;
    __bf16* Km  = (__bf16*)(ws + off); off += nX * 2;
    __bf16* Vt  = (__bf16*)(ws + off); off += nX * 2;

    cvt_bf16<<<(int)(nX / 1024), 256, 0, stream>>>(hs, Xb, (int)nX);
    cvt_bf16<<<(int)(nW / 1024), 256, 0, stream>>>(Wq, Wqb, (int)nW);
    cvt_bf16<<<(int)(nW / 1024), 256, 0, stream>>>(Wk, Wkb, (int)nW);
    cvt_bf16<<<(int)(nW / 1024), 256, 0, stream>>>(Wv, Wvb, (int)nW);

    // (B*S/16) * (HD/16) tiles = 512*48 = 24576 waves / 8 per block
    qkv_proj<<<3072, 256, 0, stream>>>(Xb, Wqb, Wkb, Wvb, bq, bk, bv, Qo, Km, Vt);

    // B*H*(S/128) blocks = 768, 8 waves each
    attn<<<768, 256, 0, stream>>>(Qo, Km, Vt, msk, out);
}
